// Attention_68307159876157
// MI455X (gfx1250) — matrix operand order repaired and checked by reading, not compiled, not run on hardware
//
#include <hip/hip_runtime.h>
#include <hip/hip_bf16.h>

#define B_    4
#define S_    2048
#define H_    2048
#define NH_   16
#define HD_   64
#define KVC_  512
#define SCALE_LOG2E 0.1803368801111244f   /* 0.125 * log2(e) */

typedef __bf16 bf16;
typedef __attribute__((ext_vector_type(16))) __bf16 v16bf;
typedef __attribute__((ext_vector_type(8)))  __bf16 v8bf;
typedef __attribute__((ext_vector_type(8)))  float  v8f;
typedef __attribute__((ext_vector_type(4)))  float  v4f;
typedef __attribute__((ext_vector_type(4)))  int    v4i;

// ---------------------------------------------------------------------------
// gfx1250 async global->LDS copy (ASYNCcnt path), with safe fallback.
// Probe-learned signature: (v4i AS1* global_src, v4i AS3* lds_dst, imm, imm)
// ---------------------------------------------------------------------------
#if defined(__gfx1250__) && __has_builtin(__builtin_amdgcn_global_load_async_to_lds_b128)
#define USE_ASYNC_LDS 1
typedef __attribute__((address_space(1))) v4i as1_v4i;
typedef __attribute__((address_space(3))) v4i as3_v4i;
#else
#define USE_ASYNC_LDS 0
#endif

__device__ __forceinline__ void copy16(const bf16* __restrict__ g, bf16* __restrict__ l) {
#if USE_ASYNC_LDS
  // flat LDS addresses carry the LDS offset in [31:0]; global AS1 == flat VA
  __builtin_amdgcn_global_load_async_to_lds_b128(
      (as1_v4i*)(unsigned long long)g,
      (as3_v4i*)(unsigned)(unsigned long long)l, 0, 0);
#else
  *(v8bf*)l = *(const v8bf*)g;
#endif
}

__device__ __forceinline__ void wait_async0() {
#if USE_ASYNC_LDS
#if __has_builtin(__builtin_amdgcn_s_wait_asynccnt)
  __builtin_amdgcn_s_wait_asynccnt(0);
#else
  asm volatile("s_wait_asynccnt 0x0" ::: "memory");
#endif
#endif
}

// ---------------------------------------------------------------------------
// fp32 -> bf16 demotion (x4 vectorized, grid-stride)
// ---------------------------------------------------------------------------
__global__ void cvt_f32_to_bf16(const float* __restrict__ s, bf16* __restrict__ d, long n) {
  long i = ((long)blockIdx.x * blockDim.x + threadIdx.x) * 4;
  long stride = (long)gridDim.x * blockDim.x * 4;
  for (; i + 3 < n; i += stride) {
    v4f f = *(const v4f*)(s + i);
    d[i + 0] = (bf16)f[0];
    d[i + 1] = (bf16)f[1];
    d[i + 2] = (bf16)f[2];
    d[i + 3] = (bf16)f[3];
  }
}

// fp32 [K][N] -> bf16 [N][K] (weights; one-time transpose so LDS B-tiles are
// K-contiguous and every WMMA B fragment is two ds_load_b128)
__global__ void cvt_transpose_bf16(const float* __restrict__ s, bf16* __restrict__ d,
                                   int K, int N) {
  long idx = (long)blockIdx.x * blockDim.x + threadIdx.x;
  long total = (long)K * N;
  long stride = (long)gridDim.x * blockDim.x;
  for (; idx < total; idx += stride) {
    long k = idx / N, n = idx % N;
    d[n * (long)K + k] = (bf16)s[idx];
  }
}

// ---------------------------------------------------------------------------
// WMMA fragment loaders (wave32, v_wmma_f32_16x16x32_bf16)
// A 16x32: lane%16 = row; element e -> k = (lane/16)*16 + e
// B 32x16: lane%16 = col, lane/16 = K half; element e -> k = base + e
// C 16x16 f32: row = vgpr + 8*(lane/16), col = lane%16
// ---------------------------------------------------------------------------
__device__ __forceinline__ v16bf load_a_frag(const bf16* p, int ld, int mBase, int kBase) {
  const int lane = threadIdx.x & 31;
  const bf16* row = p + (mBase + (lane & 15)) * ld + kBase;
  const int kh = (lane >> 4) << 4;
  v16bf a;
#pragma unroll
  for (int e = 0; e < 16; ++e)
    a[e] = row[kh + e];
  return a;
}

// B fragment from [N][K]-layout LDS tile: fully K-contiguous per lane
__device__ __forceinline__ v16bf load_bT_frag(const bf16* p, int ld, int nBase, int kBase) {
  const int lane = threadIdx.x & 31;
  const bf16* row = p + (nBase + (lane & 15)) * ld + kBase + ((lane >> 4) << 4);
  v16bf b;
#pragma unroll
  for (int e = 0; e < 16; ++e) b[e] = row[e];
  return b;
}

// B fragment from [K][N]-layout LDS tile (only used for P@V)
__device__ __forceinline__ v16bf load_b_frag(const bf16* p, int ld, int nBase, int kBase) {
  const int lane = threadIdx.x & 31;
  const int n  = nBase + (lane & 15);
  const int k0 = kBase + ((lane >> 4) << 4);
  v16bf b;
#pragma unroll
  for (int e = 0; e < 16; ++e) b[e] = p[(k0 + e) * ld + n];
  return b;
}

#define WMMA_BF16(A, Bf, C) \
  __builtin_amdgcn_wmma_f32_16x16x32_bf16(false, (A), false, (Bf), (short)0, (C), false, false)

// ---------------------------------------------------------------------------
// bf16 GEMM: C[M,N] = A[M,K] @ B[K,N] with B supplied TRANSPOSED ([N][K]).
// Block tile 128x64x64, 8 waves x (16x64), async double-buffered LDS staging,
// one barrier per K-stage. B fragments pre-loaded into distinct registers so
// the DS loads overlap the WMMA pipe (staggered dscnt waits).
// ---------------------------------------------------------------------------
template <bool OUT_F32>
__global__ __launch_bounds__(256) void gemm_bf16_bt(const bf16* __restrict__ A,
                                                    const bf16* __restrict__ BT,
                                                    void* __restrict__ Cv,
                                                    int M, int N, int K) {
  constexpr int BM = 128, BN = 64, BK = 64;
  __shared__ __align__(16) bf16 As[2][BM * BK];   // 2 x 16 KB
  __shared__ __align__(16) bf16 Bs[2][BN * BK];   // 2 x 8 KB, [N][K] layout
  const int tid  = threadIdx.x;
  const int wave = tid >> 5;
  const int lane = tid & 31;
  const int mW   = wave << 4;
  const long cm  = (long)blockIdx.y * BM;
  const long cn  = (long)blockIdx.x * BN;

  auto stage = [&](int buf, int k0) {
#pragma unroll
    for (int it = 0; it < 4; ++it) {            // A: 128x64 = 1024 x 16B
      int i = (tid + it * 256) * 8;
      int r = i >> 6, c = i & 63;
      copy16(&A[(cm + r) * K + k0 + c], &As[buf][i]);
    }
#pragma unroll
    for (int it = 0; it < 2; ++it) {            // BT: 64x64 = 512 x 16B
      int i = (tid + it * 256) * 8;
      int r = i >> 6, c = i & 63;
      copy16(&BT[(cn + r) * K + k0 + c], &Bs[buf][i]);
    }
  };

  v8f acc[4] = {};
  stage(0, 0);
  const int T = K / BK;
  for (int t = 0; t < T; ++t) {
    wait_async0();
    __syncthreads();                            // tile[t] resident for all waves
    if (t + 1 < T) stage((t + 1) & 1, (t + 1) * BK);
    const bf16* a = As[t & 1];
    const bf16* b = Bs[t & 1];
#pragma unroll
    for (int ks = 0; ks < 2; ++ks) {
      v16bf af  = load_a_frag(a, BK, mW, ks * 32);
      v16bf bf0 = load_bT_frag(b, BK,  0, ks * 32);
      v16bf bf1 = load_bT_frag(b, BK, 16, ks * 32);
      v16bf bf2 = load_bT_frag(b, BK, 32, ks * 32);
      v16bf bf3 = load_bT_frag(b, BK, 48, ks * 32);
      acc[0] = WMMA_BF16(af, bf0, acc[0]);
      acc[1] = WMMA_BF16(af, bf1, acc[1]);
      acc[2] = WMMA_BF16(af, bf2, acc[2]);
      acc[3] = WMMA_BF16(af, bf3, acc[3]);
    }
  }

  const int rHalf = (lane >> 4) << 3;
  const int nl    = lane & 15;
#pragma unroll
  for (int j = 0; j < 4; ++j) {
#pragma unroll
    for (int r = 0; r < 8; ++r) {
      long m = cm + mW + rHalf + r;
      long n = cn + j * 16 + nl;
      if (OUT_F32) ((float*)Cv)[m * (long)N + n] = acc[j][r];
      else         ((bf16*)Cv)[m * (long)N + n] = (bf16)acc[j][r];
    }
  }
}

// ---------------------------------------------------------------------------
// Causal flash attention: one (b,h) per blockIdx.y, 64 queries per blockIdx.x,
// 4 waves x 16 query rows. Async double-buffered K/V tiles, online softmax in
// C-fragment layout, P re-laid out via LDS for the P@V WMMA. Base-2 softmax.
// ---------------------------------------------------------------------------
__global__ __launch_bounds__(128) void mla_flash_attn(const bf16* __restrict__ Q,
                                                      const bf16* __restrict__ Kt,
                                                      const bf16* __restrict__ Vt,
                                                      bf16* __restrict__ O) {
  constexpr int BQ = 64, BKEY = 64, LD = NH_ * HD_;  // LD = 1024
  __shared__ __align__(16) bf16 Qs[BQ * HD_];        // 8 KB
  __shared__ __align__(16) bf16 Ks[2][BKEY * HD_];   // 2 x 8 KB
  __shared__ __align__(16) bf16 Vs[2][BKEY * HD_];   // 2 x 8 KB
  __shared__ __align__(16) bf16 Ps[BQ * BKEY];       // 8 KB

  const int tid   = threadIdx.x;
  const int wave  = tid >> 5;
  const int lane  = tid & 31;
  const int mW    = wave << 4;
  const int rHalf = (lane >> 4) << 3;
  const int nl    = lane & 15;
  const int bh    = blockIdx.y;
  const long base = ((long)(bh / NH_) * S_) * LD + (long)(bh % NH_) * HD_;
  const int q0    = blockIdx.x * BQ;

  auto stage_kv = [&](int buf, int kb) {
#pragma unroll
    for (int it = 0; it < 4; ++it) {          // 64x64 each, 512 x 16B
      int i = (tid + it * 128) * 8;
      int r = i >> 6, c = i & 63;
      long g = base + (long)(kb * BKEY + r) * LD + c;
      copy16(&Kt[g], &Ks[buf][i]);
      copy16(&Vt[g], &Vs[buf][i]);
    }
  };

#pragma unroll
  for (int it = 0; it < 4; ++it) {
    int i = (tid + it * 128) * 8;
    int r = i >> 6, c = i & 63;
    copy16(&Q[base + (long)(q0 + r) * LD + c], &Qs[i]);
  }
  stage_kv(0, 0);

  float rowMax[8], rowSum[8];
#pragma unroll
  for (int r = 0; r < 8; ++r) { rowMax[r] = -3.0e38f; rowSum[r] = 0.f; }
  v8f oacc[4] = {};

  const int nKB = blockIdx.x + 1;                    // causal
  for (int kb = 0; kb < nKB; ++kb) {
    wait_async0();
    __syncthreads();                                 // Q + K/V[kb] resident
    if (kb + 1 < nKB) stage_kv((kb + 1) & 1, kb + 1);
    const bf16* ks_ = Ks[kb & 1];
    const bf16* vs_ = Vs[kb & 1];

    // S = Q @ K^T (16q x 64k per wave); K tile is [key][hd] == [N][K]
    v8f sc[4] = {};
#pragma unroll
    for (int ks = 0; ks < 2; ++ks) {
      v16bf aq  = load_a_frag(Qs, HD_, mW, ks * 32);
      v16bf bk0 = load_bT_frag(ks_, HD_,  0, ks * 32);
      v16bf bk1 = load_bT_frag(ks_, HD_, 16, ks * 32);
      v16bf bk2 = load_bT_frag(ks_, HD_, 32, ks * 32);
      v16bf bk3 = load_bT_frag(ks_, HD_, 48, ks * 32);
      sc[0] = WMMA_BF16(aq, bk0, sc[0]);
      sc[1] = WMMA_BF16(aq, bk1, sc[1]);
      sc[2] = WMMA_BF16(aq, bk2, sc[2]);
      sc[3] = WMMA_BF16(aq, bk3, sc[3]);
    }

    // base-2 scaled scores + causal mask (diagonal block only)
    const int qRow0 = q0 + mW + rHalf;
    const bool diag = (kb == nKB - 1);
#pragma unroll
    for (int j = 0; j < 4; ++j) {
      int key = kb * BKEY + j * 16 + nl;
#pragma unroll
      for (int r = 0; r < 8; ++r) {
        float v = sc[j][r] * SCALE_LOG2E;
        if (diag && key > qRow0 + r) v = -3.0e38f;
        sc[j][r] = v;
      }
    }

    // online softmax, per-row reduction over the 16-lane half
#pragma unroll
    for (int r = 0; r < 8; ++r) {
      float m = fmaxf(fmaxf(sc[0][r], sc[1][r]), fmaxf(sc[2][r], sc[3][r]));
#pragma unroll
      for (int off = 8; off >= 1; off >>= 1) m = fmaxf(m, __shfl_xor(m, off, 32));
      float mNew = fmaxf(rowMax[r], m);
      float corr = exp2f(rowMax[r] - mNew);
      rowMax[r] = mNew;
      float ls = 0.f;
#pragma unroll
      for (int j = 0; j < 4; ++j) {
        float p = exp2f(sc[j][r] - mNew);
        sc[j][r] = p;
        ls += p;
      }
#pragma unroll
      for (int off = 8; off >= 1; off >>= 1) ls += __shfl_xor(ls, off, 32);
      rowSum[r] = rowSum[r] * corr + ls;
#pragma unroll
      for (int j = 0; j < 4; ++j) oacc[j][r] *= corr;
    }

    // C-layout -> row-major bf16 P tile
#pragma unroll
    for (int j = 0; j < 4; ++j)
#pragma unroll
      for (int r = 0; r < 8; ++r)
        Ps[(mW + rHalf + r) * BKEY + j * 16 + nl] = (bf16)sc[j][r];
    __syncthreads();                                 // Ps visible

    // O += P @ V ; V tile is [key][hd] == [K][N]
#pragma unroll
    for (int ks = 0; ks < 2; ++ks) {
      v16bf ap  = load_a_frag(Ps, BKEY, mW, ks * 32);
      v16bf bv0 = load_b_frag(vs_, HD_,  0, ks * 32);
      v16bf bv1 = load_b_frag(vs_, HD_, 16, ks * 32);
      v16bf bv2 = load_b_frag(vs_, HD_, 32, ks * 32);
      v16bf bv3 = load_b_frag(vs_, HD_, 48, ks * 32);
      oacc[0] = WMMA_BF16(ap, bv0, oacc[0]);
      oacc[1] = WMMA_BF16(ap, bv1, oacc[1]);
      oacc[2] = WMMA_BF16(ap, bv2, oacc[2]);
      oacc[3] = WMMA_BF16(ap, bv3, oacc[3]);
    }
  }

  // normalize + store (strided per-head layout [B*S, NH*HD])
#pragma unroll
  for (int j = 0; j < 4; ++j) {
#pragma unroll
    for (int r = 0; r < 8; ++r) {
      long g = base + (long)(q0 + mW + rHalf + r) * LD + j * 16 + nl;
      O[g] = (bf16)(oacc[j][r] / rowSum[r]);
    }
  }
}

// ---------------------------------------------------------------------------
extern "C" void kernel_launch(void* const* d_in, const int* in_sizes, int n_in,
                              void* d_out, int out_size, void* d_ws, size_t ws_size,
                              hipStream_t stream) {
  (void)in_sizes; (void)n_in; (void)out_size; (void)ws_size;
  const float* hs   = (const float*)d_in[0];
  const float* wq   = (const float*)d_in[1];
  const float* wkvd = (const float*)d_in[2];
  const float* wku  = (const float*)d_in[3];
  const float* wvu  = (const float*)d_in[4];
  const float* wout = (const float*)d_in[5];
  float* out = (float*)d_out;

  char* ws = (char*)d_ws;
  size_t off = 0;
  auto take = [&](size_t elems) -> bf16* {
    off = (off + 255) & ~(size_t)255;
    bf16* p = (bf16*)(ws + off);
    off += elems * sizeof(bf16);
    return p;
  };
  const long MT = (long)B_ * S_;                 // 8192 tokens
  const int  D  = NH_ * HD_;                     // 1024
  bf16* hsB    = take((size_t)MT * H_);
  bf16* wqT    = take((size_t)H_ * D);           // [1024][2048]
  bf16* wkvdT  = take((size_t)H_ * KVC_);        // [512][2048]
  bf16* wkuT   = take((size_t)KVC_ * D);         // [1024][512]
  bf16* wvuT   = take((size_t)KVC_ * D);         // [1024][512]
  bf16* woutT  = take((size_t)D * H_);           // [2048][1024]
  bf16* qB     = take((size_t)MT * D);
  bf16* kvB    = take((size_t)MT * KVC_);
  bf16* kB     = take((size_t)MT * D);
  bf16* vB     = take((size_t)MT * D);
  bf16* aoB    = take((size_t)MT * D);

  {
    long n = MT * (long)H_;
    long blocks = (n / 4 + 255) / 256; if (blocks > 4096) blocks = 4096;
    cvt_f32_to_bf16<<<(int)blocks, 256, 0, stream>>>(hs, hsB, n);
  }
  cvt_transpose_bf16<<<2048, 256, 0, stream>>>(wq,   wqT,   H_,   D);
  cvt_transpose_bf16<<<2048, 256, 0, stream>>>(wkvd, wkvdT, H_,   KVC_);
  cvt_transpose_bf16<<<1024, 256, 0, stream>>>(wku,  wkuT,  KVC_, D);
  cvt_transpose_bf16<<<1024, 256, 0, stream>>>(wvu,  wvuT,  KVC_, D);
  cvt_transpose_bf16<<<2048, 256, 0, stream>>>(wout, woutT, D,    H_);

  const dim3 blk(256);
  // q = hidden @ wq
  gemm_bf16_bt<false><<<dim3(D / 64, 8192 / 128), blk, 0, stream>>>(hsB, wqT, qB, 8192, D, H_);
  // kv_latent = hidden @ w_kv_down
  gemm_bf16_bt<false><<<dim3(KVC_ / 64, 8192 / 128), blk, 0, stream>>>(hsB, wkvdT, kvB, 8192, KVC_, H_);
  // k = kv_latent @ w_k_up
  gemm_bf16_bt<false><<<dim3(D / 64, 8192 / 128), blk, 0, stream>>>(kvB, wkuT, kB, 8192, D, KVC_);
  // v = kv_latent @ w_v_up
  gemm_bf16_bt<false><<<dim3(D / 64, 8192 / 128), blk, 0, stream>>>(kvB, wvuT, vB, 8192, D, KVC_);
  // causal flash attention per (b,h)
  mla_flash_attn<<<dim3(S_ / 64, B_ * NH_), dim3(128), 0, stream>>>(qB, kB, vB, aoB);
  // final projection -> fp32 output
  gemm_bf16_bt<true><<<dim3(H_ / 64, 8192 / 128), blk, 0, stream>>>(aoB, woutT, out, 8192, H_, D);
}
